// GCNNet_58969900974479
// MI455X (gfx1250) — compile-verified, matrix operand order checked by reading, not run on hardware
//
#include <hip/hip_runtime.h>

#define N_NODES 100000
#define N_EDGES 250000
#define B_GRAPH 2048

typedef __attribute__((ext_vector_type(16))) __bf16 v16bf;
typedef __attribute__((ext_vector_type(8)))  float  v8f;

__device__ __forceinline__ unsigned short f2bf(float f) {
    unsigned int u = __float_as_uint(f);
    u += 0x7FFFu + ((u >> 16) & 1u);          // round-to-nearest-even
    return (unsigned short)(u >> 16);
}
__device__ __forceinline__ float lrelu01(float x) { return x > 0.0f ? x : 0.01f * x; }

// ---------------- generic helpers ----------------

__global__ void k_f32_to_bf16(const float* __restrict__ src,
                              unsigned short* __restrict__ dst, int n) {
    int i = blockIdx.x * blockDim.x + threadIdx.x;
    if (i < n) dst[i] = f2bf(src[i]);
}

// W is [K, Nn] row-major f32 -> Wt is [Nn, K] row-major bf16
__global__ void k_transpose_bf16(const float* __restrict__ W,
                                 unsigned short* __restrict__ Wt, int K, int Nn) {
    int i = blockIdx.x * blockDim.x + threadIdx.x;
    if (i < K * Nn) {
        int k = i / Nn, n = i % Nn;
        Wt[(size_t)n * K + k] = f2bf(W[i]);
    }
}

__global__ void k_fill(float* __restrict__ p, float val, int n) {
    int i = blockIdx.x * blockDim.x + threadIdx.x;
    if (i < n) p[i] = val;
}

// ---------------- graph preprocessing ----------------

__global__ void k_deg_init(float* __restrict__ deg) {
    int i = blockIdx.x * blockDim.x + threadIdx.x;
    if (i < N_NODES) deg[i] = 1.0f;            // self-loop weight 1
}
__global__ void k_deg_acc(const int* __restrict__ dst, const float* __restrict__ ew,
                          float* __restrict__ deg) {
    int e = blockIdx.x * blockDim.x + threadIdx.x;
    if (e < N_EDGES) atomicAdd(&deg[dst[e]], ew[e]);
}
__global__ void k_dinv(float* __restrict__ deg) {
    int i = blockIdx.x * blockDim.x + threadIdx.x;
    if (i < N_NODES) {
        float d = deg[i];
        deg[i] = (d > 0.0f) ? rsqrtf(fmaxf(d, 1e-12f)) : 0.0f;   // in-place -> dinv
    }
}
__global__ void k_norm(const int* __restrict__ src, const int* __restrict__ dst,
                       const float* __restrict__ ew, const float* __restrict__ dinv,
                       float* __restrict__ nrm) {
    int e = blockIdx.x * blockDim.x + threadIdx.x;
    if (e < N_EDGES) nrm[e] = dinv[src[e]] * ew[e] * dinv[dst[e]];
}

// ---- WMMA bf16 GEMM: C[M,Nn] = A[M,K] * Bt[Nn,K]^T (+bias/act, +row-scaled aux) ----
// One wave computes a 16x64 strip (4 accumulators): A chunk loaded once per K-step
// and reused across 4 WMMAs. K is a compile-time constant so every global load in
// the inner loop becomes base + immediate offset (no per-iteration address VALU).
// K % 32 == 0, M % 16 == 0, Nn % 64 == 0.
// CDNA5 16-bit A 16x32 layout: lane half 0 holds K dwords {0..3, 8..11} of each
// 32-K chunk, half 1 holds {4..7, 12..15}; B symmetric with N = lane%16 (hence
// Bt stored transposed so each B column is a contiguous row).

template <int K, bool ACT, bool BF, bool AUX>
__global__ void k_wmma_gemm(const unsigned short* __restrict__ A,
                            const unsigned short* __restrict__ Bt,
                            float* __restrict__ C,
                            unsigned short* __restrict__ Cbf,      // used iff BF
                            const float* __restrict__ bias,        // optional
                            const float* __restrict__ rowscale,    // used iff AUX
                            float* __restrict__ Caux,              // used iff AUX
                            int M, int Nn) {
    const int wavesPerBlock = blockDim.x >> 5;
    int wave = blockIdx.x * wavesPerBlock + (threadIdx.x >> 5);
    int Mt = M >> 4, Nt4 = Nn >> 6;            // 64-wide N strips
    if (wave >= Mt * Nt4) return;              // wave-uniform: EXEC stays all-ones
    int tm = wave / Nt4, tn4 = wave % Nt4;
    int lane = threadIdx.x & 31;
    int j  = lane & 15;
    int hf = lane >> 4;

    const uint4* arow = (const uint4*)(A + (size_t)(tm * 16 + j) * K);
    const uint4* brow = (const uint4*)(Bt + (size_t)(tn4 * 64 + j) * K);
    constexpr int BSTR = K * 2;                // uint4s per 16 Bt rows (16*K/8)

    v8f acc[4] = {};
    union U { uint4 q4[2]; v16bf v; };
#pragma unroll 4
    for (int k0 = 0; k0 < K; k0 += 32) {
        const int q = k0 >> 3;                 // uint4 index of this 32-K chunk
        U ua;
        ua.q4[0] = arow[q + hf];
        ua.q4[1] = arow[q + 2 + hf];
#pragma unroll
        for (int t = 0; t < 4; ++t) {
            U ub;
            ub.q4[0] = brow[t * BSTR + q + hf];
            ub.q4[1] = brow[t * BSTR + q + 2 + hf];
            acc[t] = __builtin_amdgcn_wmma_f32_16x16x32_bf16(
                false, ua.v, false, ub.v, (short)0, acc[t], false, false);
        }
    }

    float rs2[8];
#pragma unroll
    for (int r = 0; r < 8; ++r) {
        rs2[r] = 1.0f;
        if (AUX) {
            float d = rowscale[tm * 16 + r + 8 * hf];
            rs2[r] = d * d;
        }
    }
#pragma unroll
    for (int t = 0; t < 4; ++t) {
        int col = tn4 * 64 + t * 16 + j;
        float bc = bias ? bias[col] : 0.0f;
#pragma unroll
        for (int r = 0; r < 8; ++r) {
            int row = tm * 16 + r + 8 * hf;
            float v = acc[t][r] + bc;
            if (ACT) v = lrelu01(v);
            size_t o = (size_t)row * Nn + col;
            C[o] = v;
            if (BF)  Cbf[o]  = f2bf(v);
            if (AUX) Caux[o] = v * rs2[r];     // self-loop term: Hw * dinv^2
        }
    }
}

// ---------------- aggregation ----------------

__global__ void k_scatter(const int* __restrict__ src, const int* __restrict__ dst,
                          const float* __restrict__ nrm,
                          const float* __restrict__ Hw, float* __restrict__ Hagg,
                          int dout) {
    int e = blockIdx.x;
    int f = blockIdx.y * blockDim.x + threadIdx.x;
    // pull a lookahead edge's gather row toward L2 (global_prefetch)
    if (threadIdx.x == 0 && blockIdx.y == 0) {
        int e2 = e + 128;
        if (e2 < N_EDGES) __builtin_prefetch(&Hw[(size_t)src[e2] * dout], 0, 0);
    }
    int s = src[e], d = dst[e];
    float w = nrm[e];
    atomicAdd(&Hagg[(size_t)d * dout + f], Hw[(size_t)s * dout + f] * w);
}

// bias + leaky_relu + eval BatchNorm, emit f32 + bf16 activations
__global__ void k_post(const float* __restrict__ Hagg,
                       const float* __restrict__ b,  const float* __restrict__ g,
                       const float* __restrict__ be, const float* __restrict__ m,
                       const float* __restrict__ v,
                       float* __restrict__ X, unsigned short* __restrict__ Xbf, int dout) {
    int i = blockIdx.x;
    int f = blockIdx.y * blockDim.x + threadIdx.x;
    float h = lrelu01(Hagg[(size_t)i * dout + f] + b[f]);
    float y = (h - m[f]) * rsqrtf(v[f] + 1e-5f) * g[f] + be[f];
    X[(size_t)i * dout + f] = y;
    Xbf[(size_t)i * dout + f] = f2bf(y);
}

// ---------------- attention pooling ----------------

__global__ void k_gate(const float* __restrict__ X, const float* __restrict__ Wg,
                       const float* __restrict__ bg, float* __restrict__ gate) {
    int node = blockIdx.x * (blockDim.x >> 5) + (threadIdx.x >> 5);
    if (node >= N_NODES) return;
    int lane = threadIdx.x & 31;
    float s = 0.0f;
    for (int f = lane; f < 512; f += 32) s += X[(size_t)node * 512 + f] * Wg[f];
    for (int off = 16; off > 0; off >>= 1) s += __shfl_xor(s, off, 32);
    if (lane == 0) gate[node] = s + bg[0];
}

__device__ __forceinline__ void atomicMaxF(float* addr, float val) {
    unsigned int* ua = (unsigned int*)addr;
    unsigned int old = *ua;
    while (__uint_as_float(old) < val) {
        unsigned int assumed = old;
        old = atomicCAS(ua, assumed, __float_as_uint(val));
        if (old == assumed) break;
    }
}
__global__ void k_gmax(const float* __restrict__ gate, const int* __restrict__ batch,
                       float* __restrict__ gmax) {
    int i = blockIdx.x * blockDim.x + threadIdx.x;
    if (i < N_NODES) atomicMaxF(&gmax[batch[i]], gate[i]);
}
__global__ void k_expsum(const float* __restrict__ gate, const int* __restrict__ batch,
                         const float* __restrict__ gmax, float* __restrict__ a,
                         float* __restrict__ asum) {
    int i = blockIdx.x * blockDim.x + threadIdx.x;
    if (i < N_NODES) {
        float e = expf(gate[i] - gmax[batch[i]]);
        a[i] = e;
        atomicAdd(&asum[batch[i]], e);
    }
}
__global__ void k_pool(const float* __restrict__ X, const float* __restrict__ a,
                       const float* __restrict__ asum, const int* __restrict__ batch,
                       float* __restrict__ pooled) {
    int i = blockIdx.x;
    int f = blockIdx.y * blockDim.x + threadIdx.x;
    int bb = batch[i];
    float w = a[i] / asum[bb];
    atomicAdd(&pooled[(size_t)bb * 512 + f], w * X[(size_t)i * 512 + f]);
}

// ---------------- final FC3 (128 -> 1) ----------------

__global__ void k_fc3(const float* __restrict__ h2, const float* __restrict__ Wf3,
                      const float* __restrict__ bf3, float* __restrict__ out) {
    int r = blockIdx.x * (blockDim.x >> 5) + (threadIdx.x >> 5);
    if (r >= B_GRAPH) return;
    int lane = threadIdx.x & 31;
    float s = 0.0f;
    for (int f = lane; f < 128; f += 32) s += h2[(size_t)r * 128 + f] * Wf3[f];
    for (int off = 16; off > 0; off >>= 1) s += __shfl_xor(s, off, 32);
    if (lane == 0) out[r] = s + bf3[0];
}

// ---------------- host orchestration ----------------

extern "C" void kernel_launch(void* const* d_in, const int* in_sizes, int n_in,
                              void* d_out, int out_size, void* d_ws, size_t ws_size,
                              hipStream_t stream) {
    const int N = N_NODES, E = N_EDGES, B = B_GRAPH;
    const int dims[6] = {64, 512, 256, 128, 256, 512};

    const float* x     = (const float*)d_in[0];
    const int*   src   = (const int*)d_in[1];          // edge_index row 0
    const int*   dst   = (const int*)d_in[1] + E;      // edge_index row 1
    const float* ea    = (const float*)d_in[2];
    const int*   batch = (const int*)d_in[3];
    const float* Wg  = (const float*)d_in[34];
    const float* bg  = (const float*)d_in[35];
    const float* Wf1 = (const float*)d_in[36];
    const float* bf1 = (const float*)d_in[37];
    const float* Wf2 = (const float*)d_in[38];
    const float* bf2 = (const float*)d_in[39];
    const float* Wf3 = (const float*)d_in[40];
    const float* bf3 = (const float*)d_in[41];
    float* out = (float*)d_out;

    // workspace carve-out
    char* p = (char*)d_ws;
    auto take = [&](size_t bytes) -> void* {
        void* r = (void*)p;
        p += (bytes + 255) & ~(size_t)255;
        return r;
    };
    float*          Xf    = (float*)take((size_t)N * 512 * 4);
    unsigned short* Xbf   = (unsigned short*)take((size_t)N * 512 * 2);
    float*          Hw    = (float*)take((size_t)N * 512 * 4);
    float*          Hagg  = (float*)take((size_t)N * 512 * 4);
    float*          dinv  = (float*)take((size_t)N * 4);
    float*          nrm   = (float*)take((size_t)E * 4);
    float*          gate  = (float*)take((size_t)N * 4);
    float*          attn  = (float*)take((size_t)N * 4);
    float*          gmax  = (float*)take((size_t)B * 4);
    float*          asum  = (float*)take((size_t)B * 4);
    float*          pooled    = (float*)take((size_t)B * 512 * 4);
    unsigned short* pooledbf  = (unsigned short*)take((size_t)B * 512 * 2);
    float*          h1    = (float*)take((size_t)B * 256 * 4);
    unsigned short* h1bf  = (unsigned short*)take((size_t)B * 256 * 2);
    float*          h2    = (float*)take((size_t)B * 128 * 4);
    unsigned short* Wt[7];
    for (int l = 0; l < 5; ++l) Wt[l] = (unsigned short*)take((size_t)dims[l] * dims[l + 1] * 2);
    Wt[5] = (unsigned short*)take((size_t)512 * 256 * 2);   // Wf1^T
    Wt[6] = (unsigned short*)take((size_t)256 * 128 * 2);   // Wf2^T

    // weights -> transposed bf16
    for (int l = 0; l < 5; ++l) {
        const float* W = (const float*)d_in[4 + l * 6];
        int n = dims[l] * dims[l + 1];
        k_transpose_bf16<<<(n + 255) / 256, 256, 0, stream>>>(W, Wt[l], dims[l], dims[l + 1]);
    }
    k_transpose_bf16<<<(512 * 256 + 255) / 256, 256, 0, stream>>>(Wf1, Wt[5], 512, 256);
    k_transpose_bf16<<<(256 * 128 + 255) / 256, 256, 0, stream>>>(Wf2, Wt[6], 256, 128);

    // degree / norm
    k_deg_init<<<(N + 255) / 256, 256, 0, stream>>>(dinv);
    k_deg_acc<<<(E + 255) / 256, 256, 0, stream>>>(dst, ea, dinv);
    k_dinv<<<(N + 255) / 256, 256, 0, stream>>>(dinv);
    k_norm<<<(E + 255) / 256, 256, 0, stream>>>(src, dst, ea, dinv, nrm);

    // input features -> bf16
    k_f32_to_bf16<<<(N * 64 + 255) / 256, 256, 0, stream>>>(x, Xbf, N * 64);

    // 5 GCN conv layers (GEMM fuses the self-loop epilogue: Hw and Hagg = Hw*dinv^2)
    for (int l = 0; l < 5; ++l) {
        int din = dims[l], dout = dims[l + 1];
        const float* b  = (const float*)d_in[4 + l * 6 + 1];
        const float* g  = (const float*)d_in[4 + l * 6 + 2];
        const float* be = (const float*)d_in[4 + l * 6 + 3];
        const float* m  = (const float*)d_in[4 + l * 6 + 4];
        const float* v  = (const float*)d_in[4 + l * 6 + 5];

        int tiles = (N / 16) * (dout / 64);
        int blocks = (tiles + 7) / 8;
        switch (din) {
        case 64:
            k_wmma_gemm<64, false, false, true><<<blocks, 256, 0, stream>>>(
                Xbf, Wt[l], Hw, nullptr, nullptr, dinv, Hagg, N, dout);
            break;
        case 128:
            k_wmma_gemm<128, false, false, true><<<blocks, 256, 0, stream>>>(
                Xbf, Wt[l], Hw, nullptr, nullptr, dinv, Hagg, N, dout);
            break;
        case 256:
            k_wmma_gemm<256, false, false, true><<<blocks, 256, 0, stream>>>(
                Xbf, Wt[l], Hw, nullptr, nullptr, dinv, Hagg, N, dout);
            break;
        default:
            k_wmma_gemm<512, false, false, true><<<blocks, 256, 0, stream>>>(
                Xbf, Wt[l], Hw, nullptr, nullptr, dinv, Hagg, N, dout);
            break;
        }

        dim3 gN(N, dout / 128), gE(E, dout / 128);
        k_scatter<<<gE, 128, 0, stream>>>(src, dst, nrm, Hw, Hagg, dout);
        k_post<<<gN, 128, 0, stream>>>(Hagg, b, g, be, m, v, Xf, Xbf, dout);
    }

    // global attention pooling over 512-wide final features
    k_gate<<<(N + 3) / 4, 128, 0, stream>>>(Xf, Wg, bg, gate);
    k_fill<<<(B + 255) / 256, 256, 0, stream>>>(gmax, -__builtin_inff(), B);
    k_fill<<<(B + 255) / 256, 256, 0, stream>>>(asum, 0.0f, B);
    k_fill<<<(B * 512 + 255) / 256, 256, 0, stream>>>(pooled, 0.0f, B * 512);
    k_gmax<<<(N + 255) / 256, 256, 0, stream>>>(gate, batch, gmax);
    k_expsum<<<(N + 255) / 256, 256, 0, stream>>>(gate, batch, gmax, attn, asum);
    k_pool<<<dim3(N, 4), 128, 0, stream>>>(Xf, attn, asum, batch, pooled);

    // FC head
    k_f32_to_bf16<<<(B * 512 + 255) / 256, 256, 0, stream>>>(pooled, pooledbf, B * 512);
    int t1 = (B / 16) * (256 / 64);
    k_wmma_gemm<512, true, true, false><<<(t1 + 7) / 8, 256, 0, stream>>>(
        pooledbf, Wt[5], h1, h1bf, bf1, nullptr, nullptr, B, 256);
    int t2 = (B / 16) * (128 / 64);
    k_wmma_gemm<256, true, false, false><<<(t2 + 7) / 8, 256, 0, stream>>>(
        h1bf, Wt[6], h2, nullptr, bf2, nullptr, nullptr, B, 128);
    k_fc3<<<(B + 3) / 4, 128, 0, stream>>>(h2, Wf3, bf3, out);
}
